// DeformableTransformerGroupDecoderLayer_49727131353263
// MI455X (gfx1250) — compile-verified, matrix operand order checked by reading, not run on hardware
//
#include <hip/hip_runtime.h>
#include <hip/hip_bf16.h>
#include <math.h>

typedef _Float16 v16h __attribute__((ext_vector_type(16)));
typedef _Float16 v8h  __attribute__((ext_vector_type(8)));
typedef float    v8f  __attribute__((ext_vector_type(8)));

#define NQc 100
#define BSc 8
#define NPc 17
#define Dc  256
#define NHc 8
#define DHc 32
#define LLc 4
#define PPc 4
#define DFFc 1024
#define T_TOK (NQc*BSc*NPc)      /* 13600 tokens */
#define LEN_INc 11253
#define VROWSc (BSc*LEN_INc)     /* 90024 value rows */

/* ------------------------------------------------------------------ */
/* f32 -> f16 one-pass conversion (weights once, activations per use) */
/* ------------------------------------------------------------------ */
__global__ void cvt_f32_f16(const float* __restrict__ a, _Float16* __restrict__ o, int n)
{
    int i = blockIdx.x * blockDim.x + threadIdx.x;
    if (i < n) o[i] = (_Float16)a[i];
}

/* ------------------------------------------------------------------ */
/* Tiled GEMM: C[M,N] = A[M,K] @ B[N,K]^T + bias                      */
/* f16 operands staged global->LDS via GLOBAL_LOAD_ASYNC_TO_LDS_B128, */
/* f32 accumulation in v_wmma_f32_16x16x32_f16.                        */
/* block = 128 threads (4 waves), tile 64x64, K-step 32               */
/* ------------------------------------------------------------------ */
#define TM 64
#define TN 64
#define TK 32
#define GEMM_THREADS 128
#define LDS_STRIDE 48   /* halfs; 96B row stride keeps 16B alignment  */

__global__ __launch_bounds__(GEMM_THREADS)
void gemm_wmma_f16(const _Float16* __restrict__ A16, const _Float16* __restrict__ B16,
                   const float* __restrict__ bias, float* __restrict__ C,
                   int M, int N, int K, int relu,
                   const unsigned char* __restrict__ row_mask)
{
    __shared__ __align__(16) _Float16 As[TM][LDS_STRIDE];
    __shared__ __align__(16) _Float16 Bs[TN][LDS_STRIDE];

    const int tid  = threadIdx.x;
    const int lane = tid & 31;
    const int wave = tid >> 5;
    const int wm   = wave >> 1;     /* 0..1 : wave row in 2x2 grid   */
    const int wn   = wave & 1;      /* 0..1 : wave col               */
    const int m16  = lane & 15;
    const int hh   = lane >> 4;     /* half-select per ISA layout    */

    const int m0 = blockIdx.y * TM;
    const int n0 = blockIdx.x * TN;

    v8f acc[2][2] = {};

    for (int k0 = 0; k0 < K; k0 += TK) {
        __syncthreads();            /* previous fragment reads done   */

        /* async-stage A(64x32) and B(64x32) f16 tiles into LDS.
           256 16-byte chunks per tile; 128 threads x 2 chunks each.  */
        #pragma unroll
        for (int it = 0; it < 2; ++it) {
            int idx = tid + GEMM_THREADS * it;   /* 0..255            */
            int r   = idx >> 2;                  /* 0..63             */
            int ch  = (idx & 3) * 8;             /* half offset 0..24 */

            int ar = m0 + r; if (ar >= M) ar = M - 1;   /* clamp: no EXEC div */
            const _Float16* ga = A16 + (size_t)ar * K + k0 + ch;
            unsigned int la = (unsigned int)(uintptr_t)&As[r][ch];
            asm volatile("global_load_async_to_lds_b128 %0, %1, off"
                         :: "v"(la), "v"((unsigned long long)(uintptr_t)ga)
                         : "memory");

            const _Float16* gb = B16 + (size_t)(n0 + r) * K + k0 + ch;  /* N % 64 == 0 */
            unsigned int lb = (unsigned int)(uintptr_t)&Bs[r][ch];
            asm volatile("global_load_async_to_lds_b128 %0, %1, off"
                         :: "v"(lb), "v"((unsigned long long)(uintptr_t)gb)
                         : "memory");
        }
        asm volatile("s_wait_asynccnt 0x0" ::: "memory");
        __syncthreads();

        /* fragments: lanes 0-15 rows, half hh selects K groups */
        v16h af[2], bf[2];
        #pragma unroll
        for (int s = 0; s < 2; ++s) {
            int row = wm * 32 + s * 16 + m16;
            v8h lo = *(const v8h*)&As[row][hh * 8];
            v8h hi = *(const v8h*)&As[row][16 + hh * 8];
            #pragma unroll
            for (int i = 0; i < 8; ++i) { af[s][i] = lo[i]; af[s][8 + i] = hi[i]; }
        }
        #pragma unroll
        for (int s = 0; s < 2; ++s) {
            int row = wn * 32 + s * 16 + m16;
            v8h lo = *(const v8h*)&Bs[row][hh * 8];
            v8h hi = *(const v8h*)&Bs[row][16 + hh * 8];
            #pragma unroll
            for (int i = 0; i < 8; ++i) { bf[s][i] = lo[i]; bf[s][8 + i] = hi[i]; }
        }
        #pragma unroll
        for (int i = 0; i < 2; ++i)
            #pragma unroll
            for (int j = 0; j < 2; ++j)
                acc[i][j] = __builtin_amdgcn_wmma_f32_16x16x32_f16(
                    false, af[i], false, bf[j], (short)0, acc[i][j], false, false);
    }

    /* epilogue: C layout per ISA (lane%16 = col, vgpr r = row within 8) */
    #pragma unroll
    for (int i = 0; i < 2; ++i) {
        #pragma unroll
        for (int j = 0; j < 2; ++j) {
            int col = n0 + wn * 32 + j * 16 + m16;
            float bb = bias ? bias[col] : 0.f;
            #pragma unroll
            for (int r = 0; r < 8; ++r) {
                int row = m0 + wm * 32 + i * 16 + hh * 8 + r;
                if (row < M) {
                    float v = acc[i][j][r] + bb;
                    if (relu) v = v > 0.f ? v : 0.f;
                    if (row_mask && row_mask[row]) v = 0.f;
                    C[(size_t)row * N + col] = v;
                }
            }
        }
    }
}

/* ------------------------------------------------------------------ */
/* Small-seq MHA core: one thread per (batch, head, query row).       */
/* 3-pass softmax with dot recompute (S<=100, DH=32).                 */
/* ------------------------------------------------------------------ */
__global__ void mha_attn(const float* __restrict__ Q, int ldq,
                         const float* __restrict__ Kp, int ldk,
                         const float* __restrict__ V, int ldv,
                         float* __restrict__ O, int ldo,
                         int nbatch, int S, int batch_stride, int seq_stride)
{
    int idx = blockIdx.x * blockDim.x + threadIdx.x;
    if (idx >= nbatch * NHc * S) return;
    int qi = idx % S;
    int h  = (idx / S) % NHc;
    int g  = idx / (S * NHc);
    int rowq = g * batch_stride + qi * seq_stride;

    const float* qp = Q + (size_t)rowq * ldq + h * DHc;
    float qr[DHc];
    #pragma unroll
    for (int d = 0; d < DHc; ++d) qr[d] = qp[d];

    const float scale = 0.17677669529663687f; /* 1/sqrt(32) */

    float mx = -3.4e38f;
    for (int kj = 0; kj < S; ++kj) {
        const float* kp = Kp + (size_t)(g * batch_stride + kj * seq_stride) * ldk + h * DHc;
        float dt = 0.f;
        #pragma unroll
        for (int d = 0; d < DHc; ++d) dt += qr[d] * kp[d];
        dt *= scale;
        mx = dt > mx ? dt : mx;
    }
    float den = 0.f;
    for (int kj = 0; kj < S; ++kj) {
        const float* kp = Kp + (size_t)(g * batch_stride + kj * seq_stride) * ldk + h * DHc;
        float dt = 0.f;
        #pragma unroll
        for (int d = 0; d < DHc; ++d) dt += qr[d] * kp[d];
        den += __expf(dt * scale - mx);
    }
    float accv[DHc];
    #pragma unroll
    for (int d = 0; d < DHc; ++d) accv[d] = 0.f;
    for (int kj = 0; kj < S; ++kj) {
        int rk = g * batch_stride + kj * seq_stride;
        const float* kp = Kp + (size_t)rk * ldk + h * DHc;
        float dt = 0.f;
        #pragma unroll
        for (int d = 0; d < DHc; ++d) dt += qr[d] * kp[d];
        float w = __expf(dt * scale - mx);
        const float* vp = V + (size_t)rk * ldv + h * DHc;
        #pragma unroll
        for (int d = 0; d < DHc; ++d) accv[d] += w * vp[d];
    }
    float inv = 1.f / den;
    float* op = O + (size_t)rowq * ldo + h * DHc;
    #pragma unroll
    for (int d = 0; d < DHc; ++d) op[d] = accv[d] * inv;
}

/* ------------------------------------------------------------------ */
__global__ void add_vec(const float* __restrict__ a, const float* __restrict__ b,
                        float* __restrict__ o, int n)
{
    int i = blockIdx.x * blockDim.x + threadIdx.x;
    if (i < n) o[i] = a[i] + b[i];
}

/* out = LayerNorm(a + b) * g + beta ; one block per token row (D=256) */
__global__ void add_ln(const float* __restrict__ A, const float* __restrict__ Bv,
                       const float* __restrict__ g, const float* __restrict__ be,
                       float* __restrict__ O)
{
    __shared__ float red[Dc];
    int row = blockIdx.x, c = threadIdx.x;
    float v = A[(size_t)row * Dc + c] + Bv[(size_t)row * Dc + c];
    red[c] = v; __syncthreads();
    for (int s = Dc / 2; s > 0; s >>= 1) { if (c < s) red[c] += red[c + s]; __syncthreads(); }
    float mean = red[0] * (1.f / Dc);
    __syncthreads();
    float dd = v - mean;
    red[c] = dd * dd; __syncthreads();
    for (int s = Dc / 2; s > 0; s >>= 1) { if (c < s) red[c] += red[c + s]; __syncthreads(); }
    float var = red[0] * (1.f / Dc);
    O[(size_t)row * Dc + c] = dd * rsqrtf(var + 1e-5f) * g[c] + be[c];
}

/* softmax over L*P=16 per (token, head); layout aw[t*128 + h*16 + i] */
__global__ void aw_softmax(float* __restrict__ aw)
{
    int idx = blockIdx.x * blockDim.x + threadIdx.x;
    if (idx >= T_TOK * NHc) return;
    float* p = aw + (size_t)idx * 16;
    float mx = p[0];
    #pragma unroll
    for (int i = 1; i < 16; ++i) mx = p[i] > mx ? p[i] : mx;
    float e[16], s = 0.f;
    #pragma unroll
    for (int i = 0; i < 16; ++i) { e[i] = __expf(p[i] - mx); s += e[i]; }
    float inv = 1.f / s;
    #pragma unroll
    for (int i = 0; i < 16; ++i) p[i] = e[i] * inv;
}

/* deformable bilinear sampling: one wave lane per (token, head, dh) */
__global__ void deform_sample(const float* __restrict__ value,
                              const float* __restrict__ offb,
                              const float* __restrict__ awb,
                              const float* __restrict__ refp,
                              float* __restrict__ samp)
{
    const int cH[4] = {92, 46, 23, 12};
    const int cW[4] = {92, 46, 23, 12};
    const int cS[4] = {0, 8464, 10580, 11109};
    int gid  = blockIdx.x * blockDim.x + threadIdx.x;
    int pair = gid >> 5;
    int lane = gid & 31;
    if (pair >= T_TOK * NHc) return;
    int h = pair & 7;
    int t = pair >> 3;
    int b = (t / NPc) % BSc;

    const float* vbase = value + (size_t)b * LEN_INc * Dc + h * DHc + lane;
    float acc = 0.f;
    #pragma unroll
    for (int l = 0; l < LLc; ++l) {
        float rx = refp[(size_t)t * LLc * 2 + l * 2 + 0];
        float ry = refp[(size_t)t * LLc * 2 + l * 2 + 1];
        int Hl = cH[l], Wl = cW[l], st = cS[l];
        #pragma unroll
        for (int p = 0; p < PPc; ++p) {
            int oi = ((h * LLc + l) * PPc + p) * 2;
            float ox = offb[(size_t)t * Dc + oi];
            float oy = offb[(size_t)t * Dc + oi + 1];
            float a  = awb[(size_t)t * (NHc * LLc * PPc) + h * LLc * PPc + l * PPc + p];
            float x  = (rx + ox / (float)Wl) * Wl - 0.5f;
            float y  = (ry + oy / (float)Hl) * Hl - 0.5f;
            float x0f = floorf(x), y0f = floorf(y);
            float wx = x - x0f, wy = y - y0f;
            int x0 = (int)x0f, y0 = (int)y0f;
            float s = 0.f;
            auto tap = [&](int yi, int xi, float w) {
                if (xi >= 0 && xi < Wl && yi >= 0 && yi < Hl)
                    s += w * vbase[(size_t)(st + yi * Wl + xi) * Dc];
            };
            tap(y0,     x0,     (1.f - wy) * (1.f - wx));
            tap(y0,     x0 + 1, (1.f - wy) * wx);
            tap(y0 + 1, x0,     wy * (1.f - wx));
            tap(y0 + 1, x0 + 1, wy * wx);
            acc += a * s;
        }
    }
    samp[(size_t)t * Dc + h * DHc + lane] = acc;
}

/* ------------------------------------------------------------------ */
extern "C" void kernel_launch(void* const* d_in, const int* in_sizes, int n_in,
                              void* d_out, int out_size, void* d_ws, size_t ws_size,
                              hipStream_t stream)
{
    (void)in_sizes; (void)n_in; (void)out_size; (void)ws_size;
    const float* tgt     = (const float*)d_in[0];
    const float* qpos    = (const float*)d_in[1];
    const float* refp    = (const float*)d_in[2];
    const float* mem     = (const float*)d_in[3];
    const float* w_in_w  = (const float*)d_in[4];
    const float* b_in_w  = (const float*)d_in[5];
    const float* w_out_w = (const float*)d_in[6];
    const float* b_out_w = (const float*)d_in[7];
    const float* w_in_a  = (const float*)d_in[8];
    const float* b_in_a  = (const float*)d_in[9];
    const float* w_out_a = (const float*)d_in[10];
    const float* b_out_a = (const float*)d_in[11];
    const float* w_off   = (const float*)d_in[12];
    const float* b_off   = (const float*)d_in[13];
    const float* w_aw    = (const float*)d_in[14];
    const float* b_aw    = (const float*)d_in[15];
    const float* w_val   = (const float*)d_in[16];
    const float* b_val   = (const float*)d_in[17];
    const float* w_oc    = (const float*)d_in[18];
    const float* b_oc    = (const float*)d_in[19];
    const float* g_w  = (const float*)d_in[20];
    const float* be_w = (const float*)d_in[21];
    const float* g_a  = (const float*)d_in[22];
    const float* be_a = (const float*)d_in[23];
    const float* g1   = (const float*)d_in[24];
    const float* be1  = (const float*)d_in[25];
    const float* g2   = (const float*)d_in[26];
    const float* be2  = (const float*)d_in[27];
    const float* w1   = (const float*)d_in[28];
    const float* b1   = (const float*)d_in[29];
    const float* w2   = (const float*)d_in[30];
    const float* b2   = (const float*)d_in[31];
    const unsigned char* mask = (const unsigned char*)d_in[32];

    float* ws = (float*)d_ws;
    float* W0 = ws;                                 /* T*768 region      */
    float* W2 = ws + (size_t)T_TOK * 768;           /* T*256             */
    float* W3 = W2 + (size_t)T_TOK * 256;           /* T*256             */
    float* W4 = W3 + (size_t)T_TOK * 256;           /* T*256             */
    float* W5 = W4 + (size_t)T_TOK * 256;           /* VROWS*256 (value/ffh) */

    /* f16 regions (16B-aligned: all offsets are multiples of 4 floats) */
    _Float16* HA = (_Float16*)(W5 + (size_t)VROWSc * 256);   /* activation scratch: VROWS*256 halfs */
    _Float16* hw = HA + (size_t)VROWSc * 256;                /* weight pool */
    _Float16* h_w_in_w  = hw; hw += 768 * 256;
    _Float16* h_w_out_w = hw; hw += 256 * 256;
    _Float16* h_w_in_a  = hw; hw += 768 * 256;
    _Float16* h_w_out_a = hw; hw += 256 * 256;
    _Float16* h_w_off   = hw; hw += 256 * 256;
    _Float16* h_w_aw    = hw; hw += 128 * 256;
    _Float16* h_w_val   = hw; hw += 256 * 256;
    _Float16* h_w_oc    = hw; hw += 256 * 256;
    _Float16* h_w1      = hw; hw += 1024 * 256;
    _Float16* h_w2      = hw; hw += 256 * 1024;

    float* BUF_QK   = W0;
    float* BUF_PQK  = W0 + (size_t)T_TOK * 256;
    float* BUF_V1   = W2;
    float* BUF_ATTN1= W3;
    float* BUF_XW   = W4;
    float* BUF_QKV2 = W0;
    float* BUF_ATTN2= W2;
    float* BUF_O2   = W3;   /* becomes x_a in place */
    float* BUF_XA   = W3;
    float* BUF_QDEF = W0;
    float* BUF_OFF  = W0 + (size_t)T_TOK * 256;
    float* BUF_AW   = W0 + (size_t)T_TOK * 512;
    float* BUF_VAL  = W5;
    float* BUF_SAMP = W4;
    float* BUF_OC   = W2;
    float* BUF_X3   = W0;
    float* BUF_FFH  = W5;
    float* BUF_FF2  = W2;

    const int n_td = T_TOK * Dc;

#define CVT(src, dst, n_) \
    cvt_f32_f16<<<((n_) + 255) / 256, 256, 0, stream>>>(src, dst, n_)
#define GEMM(Ap, Bp, biasp, Cp, M_, N_, K_, relu_, mask_) \
    gemm_wmma_f16<<<dim3((N_) / TN, ((M_) + TM - 1) / TM), GEMM_THREADS, 0, stream>>>( \
        Ap, Bp, biasp, Cp, M_, N_, K_, relu_, mask_)

    /* weights: convert once */
    CVT(w_in_w,  h_w_in_w,  768 * 256);
    CVT(w_out_w, h_w_out_w, 256 * 256);
    CVT(w_in_a,  h_w_in_a,  768 * 256);
    CVT(w_out_a, h_w_out_a, 256 * 256);
    CVT(w_off,   h_w_off,   256 * 256);
    CVT(w_aw,    h_w_aw,    128 * 256);
    CVT(w_val,   h_w_val,   256 * 256);
    CVT(w_oc,    h_w_oc,    256 * 256);
    CVT(w1,      h_w1,      1024 * 256);
    CVT(w2,      h_w2,      256 * 1024);

    /* --- MHA over points (seq 17) --- */
    add_vec<<<(n_td + 255) / 256, 256, 0, stream>>>(tgt, qpos, BUF_QK, n_td);
    CVT(BUF_QK, HA, n_td);
    GEMM(HA, h_w_in_w, b_in_w, BUF_PQK, T_TOK, 512, 256, 0, nullptr);               /* Q|K */
    CVT(tgt, HA, n_td);
    GEMM(HA, h_w_in_w + 512 * 256, b_in_w + 512, BUF_V1, T_TOK, 256, 256, 0, nullptr); /* V */
    {
        int nthr = NQc * BSc * NHc * NPc;
        mha_attn<<<(nthr + 255) / 256, 256, 0, stream>>>(
            BUF_PQK, 512, BUF_PQK + 256, 512, BUF_V1, 256, BUF_ATTN1, 256,
            NQc * BSc, NPc, NPc, 1);
    }
    CVT(BUF_ATTN1, HA, n_td);
    GEMM(HA, h_w_out_w, b_out_w, BUF_XW, T_TOK, 256, 256, 0, nullptr);
    add_ln<<<T_TOK, Dc, 0, stream>>>(tgt, BUF_XW, g_w, be_w, BUF_XW);

    /* --- MHA over queries (seq 100, strided view) --- */
    CVT(BUF_XW, HA, n_td);
    GEMM(HA, h_w_in_a, b_in_a, BUF_QKV2, T_TOK, 768, 256, 0, nullptr);
    {
        int nthr = (BSc * NPc) * NHc * NQc;
        mha_attn<<<(nthr + 255) / 256, 256, 0, stream>>>(
            BUF_QKV2, 768, BUF_QKV2 + 256, 768, BUF_QKV2 + 512, 768, BUF_ATTN2, 256,
            BSc * NPc, NQc, 1, BSc * NPc);
    }
    CVT(BUF_ATTN2, HA, n_td);
    GEMM(HA, h_w_out_a, b_out_a, BUF_O2, T_TOK, 256, 256, 0, nullptr);
    add_ln<<<T_TOK, Dc, 0, stream>>>(BUF_XW, BUF_O2, g_a, be_a, BUF_XA);

    /* --- deformable attention --- */
    add_vec<<<(n_td + 255) / 256, 256, 0, stream>>>(BUF_XA, qpos, BUF_QDEF, n_td);
    CVT(BUF_QDEF, HA, n_td);
    GEMM(HA, h_w_off, b_off, BUF_OFF, T_TOK, 256, 256, 0, nullptr);
    GEMM(HA, h_w_aw, b_aw, BUF_AW, T_TOK, 128, 256, 0, nullptr);
    aw_softmax<<<(T_TOK * NHc + 255) / 256, 256, 0, stream>>>(BUF_AW);
    CVT(mem, HA, VROWSc * Dc);
    GEMM(HA, h_w_val, b_val, BUF_VAL, VROWSc, 256, 256, 0, mask);
    {
        int nthr = T_TOK * NHc * DHc;
        deform_sample<<<(nthr + 255) / 256, 256, 0, stream>>>(
            BUF_VAL, BUF_OFF, BUF_AW, refp, BUF_SAMP);
    }
    CVT(BUF_SAMP, HA, n_td);
    GEMM(HA, h_w_oc, b_oc, BUF_OC, T_TOK, 256, 256, 0, nullptr);
    add_ln<<<T_TOK, Dc, 0, stream>>>(BUF_XA, BUF_OC, g1, be1, BUF_X3);

    /* --- FFN --- */
    CVT(BUF_X3, HA, n_td);
    GEMM(HA, h_w1, b1, BUF_FFH, T_TOK, DFFc, 256, 1, nullptr);
    CVT(BUF_FFH, HA, T_TOK * DFFc);
    GEMM(HA, h_w2, b2, BUF_FF2, T_TOK, 256, DFFc, 0, nullptr);
    add_ln<<<T_TOK, Dc, 0, stream>>>(BUF_X3, BUF_FF2, g2, be2, (float*)d_out);

#undef CVT
#undef GEMM
}
